// PieceGNN_63780264345731
// MI455X (gfx1250) — compile-verified, matrix operand order checked by reading, not run on hardware
//
#include <hip/hip_runtime.h>

typedef __attribute__((ext_vector_type(16))) __bf16 v16bf;
typedef __attribute__((ext_vector_type(8)))  __bf16 v8bf;
typedef __attribute__((ext_vector_type(8)))  float  v8f;

#define D 128   // D_IN == D_HID == D_OUT == 128

// ---------------- degree / normalization ----------------

__global__ void k_init_deg(float* deg, int n) {
    int i = blockIdx.x * blockDim.x + threadIdx.x;
    if (i < n) deg[i] = 1.0f;                 // self-loop contributes 1
}

__global__ void k_count_deg(const int* __restrict__ dst, float* deg, int e) {
    int i = blockIdx.x * blockDim.x + threadIdx.x;
    if (i < e) atomicAdd(&deg[dst[i]], 1.0f); // exact: integer-valued floats
}

__global__ void k_rsqrt_inplace(float* deg, int n) {
    int i = blockIdx.x * blockDim.x + threadIdx.x;
    if (i < n) deg[i] = rsqrtf(deg[i]);       // deg -> deg^-1/2
}

__global__ void k_zero(float* p, long long n) {
    long long i = (long long)blockIdx.x * blockDim.x + threadIdx.x;
    if (i < n) p[i] = 0.0f;
}

// ---------------- one-time packs ----------------

// Pack W (f32 row-major [K=128, N=128]) into the exact bf16 B-fragment layout:
// Wp[((w*4 + kt)*32 + lane)*16 + j] = bf16(W[K(j), w*16 + lane%16]),
// K(j) = kt*32 + (lane>=16)*8 + j + (j>=8 ? 8 : 0).
// Then the GEMM loads a lane's whole 32-byte B operand contiguously.
__global__ void k_pack_w(const float* __restrict__ W, __bf16* __restrict__ Wp) {
    const int idx = blockIdx.x * blockDim.x + threadIdx.x;   // 0 .. 16383
    const int j   = idx & 15;
    const int L   = (idx >> 4) & 31;
    const int kt  = (idx >> 9) & 3;
    const int w   = idx >> 11;
    const int col = w * 16 + (L & 15);
    const int hi  = L >> 4;
    const int k   = kt * 32 + hi * 8 + j + ((j >= 8) ? 8 : 0);
    Wp[idx] = (__bf16)W[k * D + col];
}

// Row-major f32 -> row-major bf16 (A-side operand staging, done once per matrix).
__global__ void k_f32_to_bf16(const float* __restrict__ s, __bf16* __restrict__ d,
                              long long n) {
    long long i = (long long)blockIdx.x * blockDim.x + threadIdx.x;
    if (i < n) d[i] = (__bf16)s[i];
}

// ---------------- GEMM: H[N,128] = A[N,128] @ W[128,128] (bf16 WMMA, f32 acc) -----
//
// Block = 256 threads = 8 waves; block owns a 16-row panel, wave w owns cols
// [16w,16w+16). K loop: 4x v_wmma_f32_16x16x32_bf16.
// A fragment: two contiguous 8-bf16 runs per lane per K-step (2x b128 loads).
// B fragment: one contiguous 16-bf16 run from the packed W (2x b128 loads).

__global__ void k_gemm_wmma_bf16(const __bf16* __restrict__ Abf,
                                 const __bf16* __restrict__ Wp,
                                 float* __restrict__ H,
                                 int nrows) {
    const int lane = threadIdx.x & 31;
    const int wave = threadIdx.x >> 5;       // 0..7 -> column tile
    const int r16  = lane & 15;              // row (A) / col (C/D)
    const int hi   = lane >> 4;              // K-half selector

    const int brow = blockIdx.x * 16;
    if (brow >= nrows) return;               // uniform (N is a multiple of 16)

    const size_t arow = (size_t)(brow + r16) * D;
    const int    gc   = wave * 16 + r16;     // global output column

    v8f acc = {};

    #pragma unroll
    for (int kt = 0; kt < 4; ++kt) {
        const int kb = kt * 32 + hi * 8;
        const v8bf alo = *(const v8bf*)(Abf + arow + kb);
        const v8bf ahi = *(const v8bf*)(Abf + arow + kb + 16);
        const v16bf a = __builtin_shufflevector(alo, ahi,
                            0, 1, 2, 3, 4, 5, 6, 7, 8, 9, 10, 11, 12, 13, 14, 15);
        const v16bf b = *(const v16bf*)(Wp + (((size_t)wave * 4 + kt) * 32 + lane) * 16);
        acc = __builtin_amdgcn_wmma_f32_16x16x32_bf16(
                  /*neg_a=*/false, a, /*neg_b=*/false, b,
                  /*c_mod=*/(short)0, acc, /*reuse_a=*/false, /*reuse_b=*/false);
    }

    #pragma unroll
    for (int r = 0; r < 8; ++r) {
        const int m = r + hi * 8;            // C/D layout: VGPR r -> M=r (+8 for hi lanes)
        H[(size_t)(brow + m) * D + gc] = acc[r];
    }
}

// ---------------- edge scatter: agg[dst] += h[src] * dinv[src]*dinv[dst] ----------
// One wave32 per edge: 32 lanes x float4 = one full 128-float feature row.

__global__ void k_scatter_edges(const float* __restrict__ h,
                                const float* __restrict__ dinv,
                                const int* __restrict__ src,
                                const int* __restrict__ dst,
                                float* agg, int e) {
    const int lane = threadIdx.x & 31;
    const int edge = blockIdx.x * (blockDim.x >> 5) + (threadIdx.x >> 5);
    if (edge >= e) return;

    const int   s   = src[edge];             // uniform per wave -> scalarized
    const int   d   = dst[edge];
    const float nrm = dinv[s] * dinv[d];

    const float4 v = ((const float4*)(h + (size_t)s * D))[lane];
    float* o = agg + (size_t)d * D + (lane << 2);
    atomicAdd(o + 0, v.x * nrm);
    atomicAdd(o + 1, v.y * nrm);
    atomicAdd(o + 2, v.z * nrm);
    atomicAdd(o + 3, v.w * nrm);
}

// ---------------- epilogues (self-loop h[i]*dinv^2 folded here) -------------------

// Layer 1: result is only ever consumed as the next GEMM's A operand -> emit bf16.
__global__ void k_epilogue_relu_bf16(const float* __restrict__ agg,
                                     const float* __restrict__ h,
                                     const float* __restrict__ dinv,
                                     const float* __restrict__ bias,
                                     __bf16* __restrict__ out,
                                     long long total) {
    long long i = (long long)blockIdx.x * blockDim.x + threadIdx.x;
    if (i >= total) return;
    const int node = (int)(i >> 7);
    const int c    = (int)(i & (D - 1));
    const float di = dinv[node];
    const float v  = agg[i] + h[i] * di * di + bias[c];
    out[i] = (__bf16)fmaxf(v, 0.0f);
}

// Layer 2: final f32 output; safe in-place on agg (elementwise read-then-write).
__global__ void k_epilogue_f32(const float* __restrict__ h,
                               const float* __restrict__ dinv,
                               const float* __restrict__ bias,
                               float* agg_out,
                               long long total) {
    long long i = (long long)blockIdx.x * blockDim.x + threadIdx.x;
    if (i >= total) return;
    const int node = (int)(i >> 7);
    const int c    = (int)(i & (D - 1));
    const float di = dinv[node];
    agg_out[i] = agg_out[i] + h[i] * di * di + bias[c];
}

// ---------------- driver ----------------

extern "C" void kernel_launch(void* const* d_in, const int* in_sizes, int n_in,
                              void* d_out, int out_size, void* d_ws, size_t ws_size,
                              hipStream_t stream) {
    const float* x  = (const float*)d_in[0];
    const int*   ei = (const int*)d_in[1];   // (2, E) int32 under default JAX config
    const float* W1 = (const float*)d_in[2];
    const float* b1 = (const float*)d_in[3];
    const float* W2 = (const float*)d_in[4];
    const float* b2 = (const float*)d_in[5];

    const int N = in_sizes[0] / D;
    const int E = in_sizes[1] / 2;
    const int* src = ei;
    const int* dst = ei + E;

    float* out = (float*)d_out;
    const long long total = (long long)N * D;

    // workspace: dinv [N f32] | bufA [N*D f32] | bufBF [N*D bf16] | W1p | W2p
    char*  ws   = (char*)d_ws;
    float* dinv = (float*)ws;
    size_t off  = ((size_t)N * sizeof(float) + 255) & ~(size_t)255;
    float* bufA = (float*)(ws + off);
    off += (size_t)N * D * sizeof(float);
    __bf16* bufBF = (__bf16*)(ws + off);
    off += (size_t)N * D * sizeof(__bf16);
    __bf16* W1p = (__bf16*)(ws + off);
    off += (size_t)D * D * sizeof(__bf16);
    __bf16* W2p = (__bf16*)(ws + off);

    const int TB = 256;
    const dim3 blkN((N + TB - 1) / TB);
    const dim3 blkE((E + TB - 1) / TB);
    const dim3 blkEdge((E + 7) / 8);                 // 8 waves (edges) per block
    const dim3 blkElem((unsigned)((total + TB - 1) / TB));
    const dim3 blkGemm((N + 15) / 16);
    const dim3 blkPack((D * D) / TB);                // 16384 elements

    // degrees -> dinv (shared by both layers)
    k_init_deg<<<blkN, TB, 0, stream>>>(dinv, N);
    k_count_deg<<<blkE, TB, 0, stream>>>(dst, dinv, E);
    k_rsqrt_inplace<<<blkN, TB, 0, stream>>>(dinv, N);

    // one-time operand staging
    k_pack_w<<<blkPack, TB, 0, stream>>>(W1, W1p);
    k_pack_w<<<blkPack, TB, 0, stream>>>(W2, W2p);
    k_f32_to_bf16<<<blkElem, TB, 0, stream>>>(x, bufBF, total);

    // ---- layer 1 ----
    k_gemm_wmma_bf16<<<blkGemm, TB, 0, stream>>>(bufBF, W1p, bufA, N); // h1
    k_zero<<<blkElem, TB, 0, stream>>>(out, total);                    // agg1 = 0
    k_scatter_edges<<<blkEdge, TB, 0, stream>>>(bufA, dinv, src, dst, out, E);
    k_epilogue_relu_bf16<<<blkElem, TB, 0, stream>>>(out, bufA, dinv, b1, bufBF, total);

    // ---- layer 2 ----
    k_gemm_wmma_bf16<<<blkGemm, TB, 0, stream>>>(bufBF, W2p, bufA, N); // h2
    k_zero<<<blkElem, TB, 0, stream>>>(out, total);                    // agg2 = 0
    k_scatter_edges<<<blkEdge, TB, 0, stream>>>(bufA, dinv, src, dst, out, E);
    k_epilogue_f32<<<blkElem, TB, 0, stream>>>(bufA, dinv, b2, out, total);
}
